// GraphTransformerNet_54400055771440
// MI455X (gfx1250) — matrix operand order repaired and checked by reading, not compiled, not run on hardware
//
#include <hip/hip_runtime.h>
#include <hip/hip_bf16.h>
#include <stdint.h>

// ---------------- problem constants (from reference) ----------------
#define N_NODES  25000
#define E_EDGES  400000
#define G_GRAPHS 128
#define HDIM     128
#define NLAYERS  4
#define ATT_SCALE 0.25f          // 1/sqrt(D_HEAD=16)
#define BN_EPS    1e-5f

// ---------------- CDNA5 async global->LDS availability ----------------
#if defined(__has_builtin)
#if __has_builtin(__builtin_amdgcn_global_load_async_to_lds_b128) && \
    __has_builtin(__builtin_amdgcn_s_wait_asynccnt)
#define USE_ASYNC 1
#endif
#endif
#ifndef USE_ASYNC
#define USE_ASYNC 0
#endif

// ---------------- WMMA types ----------------
typedef __bf16 bf16x16 __attribute__((ext_vector_type(16)));
typedef float  f32x8   __attribute__((ext_vector_type(8)));
union Frag { bf16x16 v; uint4 q[2]; };

__device__ __forceinline__ unsigned short f2bf(float f) {
  unsigned u = __float_as_uint(f);
  unsigned r = u + 0x7FFFu + ((u >> 16) & 1u);   // round-to-nearest-even
  return (unsigned short)(r >> 16);
}

#if USE_ASYNC
typedef int v4i_b128 __attribute__((vector_size(4 * sizeof(int))));
__device__ __forceinline__ void async_cp16(const void* g, void* l) {
  // 16B global -> LDS DMA, tracked by ASYNCcnt
  // signature (probe-derived): (AS1 int4* gaddr, AS3 int4* lds, int offset, int cpol)
  __builtin_amdgcn_global_load_async_to_lds_b128(
      (__attribute__((address_space(1))) v4i_b128*)g,
      (__attribute__((address_space(3))) v4i_b128*)l, 0, 0);
}
#endif

// =====================================================================
// conversion kernels: f32 -> bf16, K padded up to Kp (zero fill)
// =====================================================================
__global__ void act_cvt_k(const float* __restrict__ A, unsigned short* __restrict__ B,
                          long M, int K, int Kp) {
  long t = (long)blockIdx.x * 256 + threadIdx.x;
  if (t >= M * Kp) return;
  int k = (int)(t % Kp);
  long m = t / Kp;
  B[t] = (k < K) ? f2bf(A[m * K + k]) : (unsigned short)0;
}

// W[K x 128] f32 -> Wt[128 x Kp] bf16 (transposed, zero-padded)
__global__ void wt_cvt_k(const float* __restrict__ W, unsigned short* __restrict__ Wt,
                         int K, int Kp) {
  int t = blockIdx.x * 256 + threadIdx.x;
  if (t >= 128 * Kp) return;
  int n = t / Kp, k = t % Kp;
  Wt[t] = (k < K) ? f2bf(W[(size_t)k * 128 + n]) : (unsigned short)0;
}

// =====================================================================
// WMMA GEMM: out[M x 128] = A[M x Kp]_bf16 * Wt[128 x Kp]_bf16^T
// 64x128 block tile, 8 wave32 waves, 4x v_wmma_f32_16x16x32_bf16 / K-step.
// Double-buffered LDS; async global->LDS DMA pipeline when available.
// =====================================================================
template<bool HAS_BIAS, bool ADD_C, bool HAS_RES, bool RELU>
__global__ __launch_bounds__(256)
void gemm128_wmma(const unsigned short* __restrict__ Abf,
                  const unsigned short* __restrict__ Wt,
                  const float* __restrict__ bias, const float* __restrict__ res,
                  float* __restrict__ out, int M, int Kp) {
  constexpr int BM = 64, BK = 32;
  __shared__ __align__(16) unsigned short lA[2][BM * BK];    //  8 KB
  __shared__ __align__(16) unsigned short lB[2][HDIM * BK];  // 16 KB

  const int tid    = threadIdx.x;
  const int lane   = tid & 31;
  const int wave   = tid >> 5;
  const int rt     = wave & 3;        // row tile 0..3
  const int cgrp   = wave >> 2;       // 0 -> cols 0..63, 1 -> cols 64..127
  const int halfw  = lane >> 4;
  const int nl     = lane & 15;
  const int rowBase = blockIdx.x * BM;

  // ---- branch-free staging addresses (16B chunks) ----
  const int ar = tid >> 2, ac = (tid & 3) * 8;            // A: 64 rows x 4 chunks
  const int arow = (rowBase + ar < M) ? (rowBase + ar) : (M - 1);   // clamp: only feeds masked rows
  const unsigned short* gA = Abf + (size_t)arow * Kp + ac;
  const int aOff = ar * BK + ac;
  // B tile = 128 rows x 64B; thread stages rows tid>>2 and (tid>>2)+64
  const int br = tid >> 2, bc = (tid & 3) * 8;
  const unsigned short* gB0 = Wt + (size_t)br * Kp + bc;
  const unsigned short* gB1 = Wt + (size_t)(br + 64) * Kp + bc;
  const int bOff0 = br * BK + bc, bOff1 = (br + 64) * BK + bc;

  const int KT = Kp >> 5;
  f32x8 acc[4] = {};

  auto stage = [&](int buf, int k0) {
#if USE_ASYNC
    async_cp16(gA + k0,  &lA[buf][aOff]);
    async_cp16(gB0 + k0, &lB[buf][bOff0]);
    async_cp16(gB1 + k0, &lB[buf][bOff1]);
#else
    *(uint4*)&lA[buf][aOff]  = *(const uint4*)(gA + k0);
    *(uint4*)&lB[buf][bOff0] = *(const uint4*)(gB0 + k0);
    *(uint4*)&lB[buf][bOff1] = *(const uint4*)(gB1 + k0);
#endif
  };

  stage(0, 0);
  for (int kt = 0; kt < KT; ++kt) {
    const int cur = kt & 1;
    const bool hasNext = (kt + 1 < KT);
    if (hasNext) stage(1 - cur, (kt + 1) * BK);   // overlap next tile's DMA with this tile's math
#if USE_ASYNC
    if (hasNext) __builtin_amdgcn_s_wait_asynccnt(3);  // in-order: tile kt has landed
    else         __builtin_amdgcn_s_wait_asynccnt(0);
#endif
    __syncthreads();

    Frag af;
    af.q[0] = *(const uint4*)&lA[cur][(rt * 16 + nl) * BK + halfw * 8];
    af.q[1] = *(const uint4*)&lA[cur][(rt * 16 + nl) * BK + 16 + halfw * 8];
#pragma unroll
    for (int t = 0; t < 4; ++t) {
      const int ct = cgrp * 4 + t;
      Frag bf_;
      const uint4* pb = (const uint4*)&lB[cur][(ct * 16 + nl) * BK + halfw * 8];
      bf_.q[0] = pb[0];
      bf_.q[1] = pb[2];
      acc[t] = __builtin_amdgcn_wmma_f32_16x16x32_bf16(
          false, af.v, false, bf_.v, (short)0, acc[t], false, false);
    }
    __syncthreads();   // all reads of buf[cur] done before it is re-staged at kt+2
  }

  // ---- epilogue: C VGPR r -> row r + 8*halfw, col = lane&15 ----
#pragma unroll
  for (int t = 0; t < 4; ++t) {
    const int col = (cgrp * 4 + t) * 16 + nl;
    float bv = HAS_BIAS ? bias[col] : 0.0f;
#pragma unroll
    for (int r = 0; r < 8; ++r) {
      int row = rowBase + rt * 16 + r + 8 * halfw;
      if (row < M) {
        size_t o = (size_t)row * HDIM + col;
        float v = acc[t][r] + bv;
        if (HAS_RES) v += res[o];
        if (ADD_C)   v += out[o];
        if (RELU)    v = fmaxf(v, 0.0f);
        out[o] = v;
      }
    }
  }
}

// =====================================================================
// streaming / scatter kernels (bandwidth & atomic bound)
// =====================================================================
__global__ void fill_f32(float* p, float v, long n) {
  long i = (long)blockIdx.x * 256 + threadIdx.x;
  if (i < n) p[i] = v;
}

__global__ void qijk_score_k(const float* __restrict__ Eh, const float* __restrict__ Q,
                             const float* __restrict__ Kb, const int* __restrict__ src,
                             const int* __restrict__ dst, float* __restrict__ qjk,
                             float* __restrict__ score, int E) {
  long t = (long)blockIdx.x * 256 + threadIdx.x;
  if (t >= (long)E * 8) return;
  int ei = (int)(t >> 3), hd = (int)(t & 7);
  long be = (long)ei * HDIM + hd * 16;
  long bd = (long)dst[ei] * HDIM + hd * 16;
  long bs = (long)src[ei] * HDIM + hd * 16;
  float s = 0.0f;
#pragma unroll
  for (int j = 0; j < 16; ++j) {
    float v = Eh[be + j] * Q[bd + j] * Kb[bs + j] * ATT_SCALE;
    qjk[be + j] = v;
    s += v;
  }
  score[t] = s;
}

__device__ __forceinline__ unsigned f2key(float f) {
  unsigned u = __float_as_uint(f);
  return (u & 0x80000000u) ? ~u : (u | 0x80000000u);
}
__device__ __forceinline__ float key2f(unsigned k) {
  unsigned u = (k & 0x80000000u) ? (k & 0x7FFFFFFFu) : ~k;
  return __uint_as_float(u);
}

__global__ void seg_max_k(const float* __restrict__ score, const int* __restrict__ dst,
                          unsigned* __restrict__ mkey, int E) {
  long t = (long)blockIdx.x * 256 + threadIdx.x;
  if (t >= (long)E * 8) return;
  int ei = (int)(t >> 3), hd = (int)(t & 7);
  atomicMax(&mkey[(long)dst[ei] * 8 + hd], f2key(score[t]));
}

__global__ void exp_den_k(float* __restrict__ score, const int* __restrict__ dst,
                          const unsigned* __restrict__ mkey, float* __restrict__ den, int E) {
  long t = (long)blockIdx.x * 256 + threadIdx.x;
  if (t >= (long)E * 8) return;
  int ei = (int)(t >> 3), hd = (int)(t & 7);
  float m = key2f(mkey[(long)dst[ei] * 8 + hd]);
  float ex = __expf(score[t] - m);
  score[t] = ex;
  atomicAdd(&den[(long)dst[ei] * 8 + hd], ex);
}

__global__ void aggregate_k(const float* __restrict__ ex, const float* __restrict__ den,
                            const float* __restrict__ V, const int* __restrict__ src,
                            const int* __restrict__ dst, float* __restrict__ agg, int E) {
  long t = (long)blockIdx.x * 256 + threadIdx.x;
  if (t >= (long)E * 8) return;
  int ei = (int)(t >> 3), hd = (int)(t & 7);
  float a = ex[t] / (den[(long)dst[ei] * 8 + hd] + 1e-16f);
  long bs = (long)src[ei] * HDIM + hd * 16;
  long bd = (long)dst[ei] * HDIM + hd * 16;
#pragma unroll
  for (int j = 0; j < 16; ++j)
    atomicAdd(&agg[bd + j], a * V[bs + j]);
}

__global__ void bn_stats_k(const float* __restrict__ x, float* __restrict__ sums,
                           float* __restrict__ sqs, int M) {
  int c = threadIdx.x & 127;
  int sub = threadIdx.x >> 7;
  long base = (long)blockIdx.x * 512 + sub;
  float s = 0.0f, q = 0.0f;
  for (int i = 0; i < 512; i += 2) {
    long r = base + i;
    if (r < M) {
      float v = x[r * HDIM + c];
      s += v; q += v * v;
    }
  }
  atomicAdd(&sums[c], s);
  atomicAdd(&sqs[c], q);
}

__global__ void bn_apply_k(const float* __restrict__ x, const float* __restrict__ sums,
                           const float* __restrict__ sqs, const float* __restrict__ gamma,
                           const float* __restrict__ beta, float* __restrict__ y, int M) {
  long i = (long)blockIdx.x * 256 + threadIdx.x;
  if (i >= (long)M * HDIM) return;
  int c = (int)(i & 127);
  float mu  = sums[c] / (float)M;
  float var = sqs[c] / (float)M - mu * mu;      // biased variance == x.var(0)
  y[i] = (x[i] - mu) * rsqrtf(var + BN_EPS) * gamma[c] + beta[c];
}

__global__ void pool_k(const float* __restrict__ h, const int* __restrict__ batch,
                       float* __restrict__ g, int N) {
  long t = (long)blockIdx.x * 256 + threadIdx.x;
  if (t >= (long)N * HDIM) return;
  int n = (int)(t >> 7), c = (int)(t & 127);
  atomicAdd(&g[(long)batch[n] * HDIM + c], h[t]);
}

__global__ void head_out_k(const float* __restrict__ hidmu, const float* __restrict__ hidlv,
                           const float* __restrict__ muW2, const float* __restrict__ mub2,
                           const float* __restrict__ lvW2, const float* __restrict__ lvb2,
                           float* __restrict__ out) {
  int t = threadIdx.x;
  if (t < 128) {
    float s = mub2[0];
    for (int k = 0; k < 128; ++k) s += hidmu[t * 128 + k] * muW2[k];
    out[t] = s;
  } else {
    int r = t - 128;
    float s = lvb2[0];
    for (int k = 0; k < 128; ++k) s += hidlv[r * 128 + k] * lvW2[k];
    out[128 + r] = __expf(0.5f * s);
  }
}

// =====================================================================
// host orchestration
// =====================================================================
enum EpMode { EP_NONE, EP_ADDC, EP_BIAS, EP_BIAS_RES, EP_BIAS_RELU };

static void launch_gemm(hipStream_t st, const unsigned short* A, const unsigned short* Wt,
                        const float* bias, const float* res, float* out, int M, int Kp,
                        EpMode mode) {
  dim3 grid((M + 63) / 64), block(256);
  switch (mode) {
    case EP_NONE:      gemm128_wmma<false,false,false,false><<<grid,block,0,st>>>(A,Wt,nullptr,nullptr,out,M,Kp); break;
    case EP_ADDC:      gemm128_wmma<false,true ,false,false><<<grid,block,0,st>>>(A,Wt,nullptr,nullptr,out,M,Kp); break;
    case EP_BIAS:      gemm128_wmma<true ,false,false,false><<<grid,block,0,st>>>(A,Wt,bias,nullptr,out,M,Kp);    break;
    case EP_BIAS_RES:  gemm128_wmma<true ,false,true ,false><<<grid,block,0,st>>>(A,Wt,bias,res,out,M,Kp);        break;
    case EP_BIAS_RELU: gemm128_wmma<true ,false,false,true ><<<grid,block,0,st>>>(A,Wt,bias,nullptr,out,M,Kp);    break;
  }
}

extern "C" void kernel_launch(void* const* d_in, const int* in_sizes, int n_in,
                              void* d_out, int out_size, void* d_ws, size_t ws_size,
                              hipStream_t stream) {
  (void)in_sizes; (void)n_in; (void)out_size; (void)ws_size;
  const float* x    = (const float*)d_in[0];
  const float* ea   = (const float*)d_in[1];
  const float* pe   = (const float*)d_in[2];
  const int*   eidx = (const int*)  d_in[3];
  const int*   bat  = (const int*)  d_in[4];
  const float* Wn   = (const float*)d_in[5];
  const float* We   = (const float*)d_in[6];
  const float* Wpe  = (const float*)d_in[7];
  const float* Wqkv = (const float*)d_in[8];
  const float* bqkv = (const float*)d_in[9];
  const float* WEl  = (const float*)d_in[10];
  const float* bEl  = (const float*)d_in[11];
  const float* WOl  = (const float*)d_in[12];
  const float* bOl  = (const float*)d_in[13];
  const float* WOel = (const float*)d_in[14];
  const float* bOel = (const float*)d_in[15];
  const float* fW1  = (const float*)d_in[16];
  const float* fb1  = (const float*)d_in[17];
  const float* fW2  = (const float*)d_in[18];
  const float* fb2  = (const float*)d_in[19];
  const float* feW1 = (const float*)d_in[20];
  const float* feb1 = (const float*)d_in[21];
  const float* feW2 = (const float*)d_in[22];
  const float* feb2 = (const float*)d_in[23];
  const float* bng  = (const float*)d_in[24];
  const float* bnb  = (const float*)d_in[25];
  const float* muW1 = (const float*)d_in[26];
  const float* mub1 = (const float*)d_in[27];
  const float* muW2 = (const float*)d_in[28];
  const float* mub2 = (const float*)d_in[29];
  const float* lvW1 = (const float*)d_in[30];
  const float* lvb1 = (const float*)d_in[31];
  const float* lvW2 = (const float*)d_in[32];
  const float* lvb2 = (const float*)d_in[33];

  const int* srcI = eidx;
  const int* dstI = eidx + E_EDGES;
  float* out = (float*)d_out;

  // ---- bump-allocate workspace ----
  char* wp = (char*)d_ws;
  size_t off = 0;
  auto alloc = [&](size_t bytes) -> void* {
    void* p = (void*)(wp + off);
    off += (bytes + 255) & ~(size_t)255;
    return p;
  };
  const size_t NB = (size_t)N_NODES * HDIM * 4;
  const size_t EB = (size_t)E_EDGES * HDIM * 4;
  float* h    = (float*)alloc(NB);
  float* Qb   = (float*)alloc(NB);
  float* Kb   = (float*)alloc(NB);
  float* Vb   = (float*)alloc(NB);
  float* agg  = (float*)alloc(NB);
  float* tN1  = (float*)alloc(NB);
  float* tN2  = (float*)alloc(NB);
  float* e    = (float*)alloc(EB);
  float* qjk  = (float*)alloc(EB);             // qijk; reused as edge-FFN hidden
  float* EhB  = (float*)alloc(EB);             // Eh; reused as edge pre-activation
  float* score= (float*)alloc((size_t)E_EDGES * 8 * 4);
  float* mkey = (float*)alloc((size_t)N_NODES * 8 * 4);
  float* den  = (float*)alloc((size_t)N_NODES * 8 * 4);
  float* sums = (float*)alloc(256 * 4);
  float* g    = (float*)alloc((size_t)G_GRAPHS * HDIM * 4);
  float* hid1 = (float*)alloc((size_t)G_GRAPHS * HDIM * 4);
  float* hid2 = (float*)alloc((size_t)G_GRAPHS * HDIM * 4);
  unsigned short* nAbf = (unsigned short*)alloc((size_t)N_NODES * HDIM * 2);  // node A operand (bf16)
  unsigned short* eAbf = (unsigned short*)alloc((size_t)E_EDGES * HDIM * 2);  // edge A operand (bf16)

  // ---- transposed bf16 weight pool (~1.4 MB, lives in L2) ----
  auto wt_prep = [&](const float* W, int K, int Kp) -> unsigned short* {
    unsigned short* p = (unsigned short*)alloc((size_t)128 * Kp * 2);
    int n = 128 * Kp;
    wt_cvt_k<<<dim3((n + 255) / 256), 256, 0, stream>>>(W, p, K, Kp);
    return p;
  };
  const size_t WSZ = (size_t)HDIM * HDIM;
  unsigned short* WnT  = wt_prep(Wn, 64, 64);
  unsigned short* WpeT = wt_prep(Wpe, 16, 32);
  unsigned short* WeT  = wt_prep(We, 16, 32);
  unsigned short *WqT[NLAYERS], *WkT[NLAYERS], *WvT[NLAYERS], *WElT[NLAYERS],
                 *WOT[NLAYERS], *WOeT[NLAYERS], *fW1T[NLAYERS], *fW2T[NLAYERS],
                 *feW1T[NLAYERS], *feW2T[NLAYERS];
  for (int l = 0; l < NLAYERS; ++l) {
    WqT[l]   = wt_prep(Wqkv + (size_t)(l * 3 + 0) * WSZ, 128, 128);
    WkT[l]   = wt_prep(Wqkv + (size_t)(l * 3 + 1) * WSZ, 128, 128);
    WvT[l]   = wt_prep(Wqkv + (size_t)(l * 3 + 2) * WSZ, 128, 128);
    WElT[l]  = wt_prep(WEl  + (size_t)l * WSZ, 128, 128);
    WOT[l]   = wt_prep(WOl  + (size_t)l * WSZ, 128, 128);
    WOeT[l]  = wt_prep(WOel + (size_t)l * WSZ, 128, 128);
    fW1T[l]  = wt_prep(fW1  + (size_t)l * WSZ, 128, 128);
    fW2T[l]  = wt_prep(fW2  + (size_t)l * WSZ, 128, 128);
    feW1T[l] = wt_prep(feW1 + (size_t)l * WSZ, 128, 128);
    feW2T[l] = wt_prep(feW2 + (size_t)l * WSZ, 128, 128);
  }
  unsigned short* muW1T = wt_prep(muW1, 128, 128);
  unsigned short* lvW1T = wt_prep(lvW1, 128, 128);

  auto fill = [&](float* p, float v, long n) {
    fill_f32<<<dim3((unsigned)((n + 255) / 256)), 256, 0, stream>>>(p, v, n);
  };
  auto cvt = [&](const float* src, unsigned short* dst, long M, int K, int Kp) {
    long n = M * Kp;
    act_cvt_k<<<dim3((unsigned)((n + 255) / 256)), 256, 0, stream>>>(src, dst, M, K, Kp);
  };
  auto bn = [&](const float* xin, float* yout, int lidx, int M) {
    fill(sums, 0.0f, 256);
    bn_stats_k<<<dim3((unsigned)((M + 511) / 512)), 256, 0, stream>>>(xin, sums, sums + 128, M);
    long total = (long)M * HDIM;
    bn_apply_k<<<dim3((unsigned)((total + 255) / 256)), 256, 0, stream>>>(
        xin, sums, sums + 128, bng + (size_t)lidx * 128, bnb + (size_t)lidx * 128, yout, M);
  };

  const unsigned e8blocks = (unsigned)(((long)E_EDGES * 8 + 255) / 256);

  // ---- input embeddings ----
  cvt(x, nAbf, N_NODES, 64, 64);
  launch_gemm(stream, nAbf, WnT, nullptr, nullptr, h, N_NODES, 64, EP_NONE);
  cvt(pe, nAbf, N_NODES, 16, 32);
  launch_gemm(stream, nAbf, WpeT, nullptr, nullptr, h, N_NODES, 32, EP_ADDC);
  cvt(ea, eAbf, E_EDGES, 16, 32);
  launch_gemm(stream, eAbf, WeT, nullptr, nullptr, e, E_EDGES, 32, EP_NONE);

  for (int l = 0; l < NLAYERS; ++l) {
    const float* bq = bqkv + (size_t)(l * 3 + 0) * HDIM;
    const float* bk = bqkv + (size_t)(l * 3 + 1) * HDIM;
    const float* bv = bqkv + (size_t)(l * 3 + 2) * HDIM;

    cvt(h, nAbf, N_NODES, 128, 128);                       // one conversion feeds Q,K,V
    launch_gemm(stream, nAbf, WqT[l], bq, nullptr, Qb, N_NODES, 128, EP_BIAS);
    launch_gemm(stream, nAbf, WkT[l], bk, nullptr, Kb, N_NODES, 128, EP_BIAS);
    launch_gemm(stream, nAbf, WvT[l], bv, nullptr, Vb, N_NODES, 128, EP_BIAS);
    cvt(e, eAbf, E_EDGES, 128, 128);
    launch_gemm(stream, eAbf, WElT[l], bEl + (size_t)l * HDIM, nullptr, EhB,
                E_EDGES, 128, EP_BIAS);                    // Eh

    // attention: scores + segment softmax + aggregation
    qijk_score_k<<<e8blocks, 256, 0, stream>>>(EhB, Qb, Kb, srcI, dstI, qjk, score, E_EDGES);
    fill(mkey, 0.0f, (long)N_NODES * 8);
    fill(den,  0.0f, (long)N_NODES * 8);
    fill(agg,  0.0f, (long)N_NODES * HDIM);
    seg_max_k <<<e8blocks, 256, 0, stream>>>(score, dstI, (unsigned*)mkey, E_EDGES);
    exp_den_k <<<e8blocks, 256, 0, stream>>>(score, dstI, (const unsigned*)mkey, den, E_EDGES);
    aggregate_k<<<e8blocks, 256, 0, stream>>>(score, den, Vb, srcI, dstI, agg, E_EDGES);

    // node branch
    cvt(agg, nAbf, N_NODES, 128, 128);
    launch_gemm(stream, nAbf, WOT[l], bOl + (size_t)l * HDIM, h, tN1, N_NODES, 128, EP_BIAS_RES);
    bn(tN1, h, l * 4 + 0, N_NODES);
    cvt(h, nAbf, N_NODES, 128, 128);
    launch_gemm(stream, nAbf, fW1T[l], fb1 + (size_t)l * HDIM, nullptr, tN2, N_NODES, 128, EP_BIAS_RELU);
    cvt(tN2, nAbf, N_NODES, 128, 128);
    launch_gemm(stream, nAbf, fW2T[l], fb2 + (size_t)l * HDIM, h, tN1, N_NODES, 128, EP_BIAS_RES);
    bn(tN1, h, l * 4 + 1, N_NODES);

    // edge branch
    cvt(qjk, eAbf, E_EDGES, 128, 128);
    launch_gemm(stream, eAbf, WOeT[l], bOel + (size_t)l * HDIM, e, EhB, E_EDGES, 128, EP_BIAS_RES);
    bn(EhB, e, l * 4 + 2, E_EDGES);
    cvt(e, eAbf, E_EDGES, 128, 128);
    launch_gemm(stream, eAbf, feW1T[l], feb1 + (size_t)l * HDIM, nullptr, qjk, E_EDGES, 128, EP_BIAS_RELU);
    cvt(qjk, eAbf, E_EDGES, 128, 128);
    launch_gemm(stream, eAbf, feW2T[l], feb2 + (size_t)l * HDIM, e, EhB, E_EDGES, 128, EP_BIAS_RES);
    bn(EhB, e, l * 4 + 3, E_EDGES);
  }

  // ---- pool + heads ----
  fill(g, 0.0f, (long)G_GRAPHS * HDIM);
  pool_k<<<dim3((unsigned)(((long)N_NODES * HDIM + 255) / 256)), 256, 0, stream>>>(h, bat, g, N_NODES);
  cvt(g, nAbf, G_GRAPHS, 128, 128);
  launch_gemm(stream, nAbf, muW1T, mub1, nullptr, hid1, G_GRAPHS, 128, EP_BIAS_RELU);
  launch_gemm(stream, nAbf, lvW1T, lvb1, nullptr, hid2, G_GRAPHS, 128, EP_BIAS_RELU);
  head_out_k<<<1, 256, 0, stream>>>(hid1, hid2, muW2, mub2, lvW2, lvb2, out);
}
